// BaselineClassifier_5995774345982
// MI455X (gfx1250) — compile-verified, matrix operand order checked by reading, not run on hardware
//
#include <hip/hip_runtime.h>
#include <hip/hip_bf16.h>

typedef _Float16 f16;
typedef __attribute__((ext_vector_type(16))) _Float16 v16h;
typedef __attribute__((ext_vector_type(8)))  _Float16 v8h;
typedef __attribute__((ext_vector_type(8)))  float    v8f;
typedef __attribute__((ext_vector_type(4)))  unsigned uv4;
typedef __attribute__((ext_vector_type(4)))  int      iv4;
typedef __attribute__((ext_vector_type(8)))  int      iv8;

#define WMMA_F16(A,B,C) __builtin_amdgcn_wmma_f32_16x16x32_f16(false,(A),false,(B),(short)0,(C),false,false)

// ---- problem constants ----
constexpr int NE    = 400000;      // edges
constexpr int NN    = 50000;       // nodes
constexpr int NG    = 512;         // graphs
constexpr int ROWS  = NE + NN;     // 450000 (divisible by 16)
constexpr int TILES = ROWS / 16;   // 28125
constexpr int IN_D  = 50;          // edge feature dim
constexpr int PAD_D = 64;          // padded K for WMMA
constexpr int HID   = 128;
constexpr int NC    = 10;
constexpr int LAYERS= 3;

// ---- workspace layout (bytes) ----
constexpr size_t OFF_EA16 = 0;
constexpr size_t SZ_EA16  = (size_t)ROWS * PAD_D * 2;      // f16 edge features (padded)
constexpr size_t OFF_ZERO = OFF_EA16 + SZ_EA16;            // start of zero-initialized zone
constexpr size_t OFF_EASUM= OFF_ZERO;
constexpr size_t SZ_EASUM = (size_t)NN * IN_D * 4;
constexpr size_t OFF_DEG  = OFF_EASUM + SZ_EASUM;
constexpr size_t SZ_DEG   = (size_t)NN * 4;
constexpr size_t OFF_XA   = OFF_DEG + SZ_DEG;
constexpr size_t SZ_X     = (size_t)NN * HID * 4;
constexpr size_t OFF_XB   = OFF_XA + SZ_X;
constexpr size_t OFF_MEAN = OFF_XB + SZ_X;
constexpr size_t SZ_MEAN  = (size_t)NG * HID * 4;
constexpr size_t OFF_MAX  = OFF_MEAN + SZ_MEAN;
constexpr size_t OFF_CNT  = OFF_MAX + SZ_MEAN;
constexpr size_t SZ_CNT   = (size_t)NG * 4;
constexpr size_t ZERO_BYTES = OFF_CNT + SZ_CNT - OFF_ZERO;
constexpr size_t OFF_G    = OFF_CNT + SZ_CNT;
constexpr size_t SZ_G     = (size_t)NG * 256 * 4;
// pre-transposed f16 weight images (TDM sources)
constexpr size_t OFF_W1T  = OFF_G + SZ_G;                  // [3][128][64]  f16
constexpr size_t SZ_W1T   = (size_t)LAYERS * HID * PAD_D * 2;
constexpr size_t OFF_W2T  = OFF_W1T + SZ_W1T;              // [3][128][128] f16
constexpr size_t SZ_W2T   = (size_t)LAYERS * HID * HID * 2;
constexpr size_t OFF_WC1T = OFF_W2T + SZ_W2T;              // [128][256]    f16
constexpr size_t SZ_WC1T  = (size_t)HID * 256 * 2;
constexpr size_t OFF_WC2T = OFF_WC1T + SZ_WC1T;            // [16][128]     f16
constexpr size_t SZ_WC2T  = (size_t)16 * HID * 2;

// LDS bytes (weight slabs stored [n][k], K-contiguous)
constexpr int GEMM_LDS = (64*128 + 128*128 + 8*16*128) * 2 + 2*128*4;          // 82944
constexpr int CLS_LDS  = (256*128 + 128*16 + 8*16*128) * 2 + 128*4 + 16*4;     // 102976

// ---- order-preserving float<->uint encoding for atomic max ----
__device__ __forceinline__ unsigned fenc(float f){
  unsigned u = __float_as_uint(f);
  return (u & 0x80000000u) ? ~u : (u | 0x80000000u);
}
__device__ __forceinline__ float fdec(unsigned u){
  return (u & 0x80000000u) ? __uint_as_float(u & 0x7FFFFFFFu) : __uint_as_float(~u);
}

// ---- Tensor Data Mover: 1-row bulk copy global->LDS (D# per ISA ch.8) ----
// nbytes must be a multiple of 8; lds_dst/gsrc 8B-aligned. Issued per-wave.
__device__ __forceinline__ void tdm_load_to_lds(void* lds_dst, const void* gsrc, unsigned nbytes){
  const unsigned n8 = nbytes >> 3;                       // 8-byte elements
  const unsigned lds_addr = (unsigned)(uintptr_t)lds_dst;   // addr[31:0] = wave LDS offset
  const unsigned long long ga = (unsigned long long)(uintptr_t)gsrc;
  uv4 g0;
  g0.x = 1u;                                             // count=1; no gather/iterate/restore
  g0.y = lds_addr;                                       // D#.lds_addr (bytes)
  g0.z = (unsigned)ga;                                   // global_addr[31:0]
  g0.w = (unsigned)((ga >> 32) & 0x01ffffffull) | (2u << 30); // global_addr[56:32] | type=2
  iv8 g1;
  g1[0] = (int)(3u << 16);                               // data_size=8B; wg_mask=0; no flags
  g1[1] = (int)((n8 & 0xffffu) << 16);                   // tensor_dim0[15:0]   (bits 63:48)
  g1[2] = (int)(((n8 >> 16) & 0xffffu) | (1u << 16));    // tensor_dim0[31:16]; tensor_dim1=1
  g1[3] = (int)((n8 & 0xffffu) << 16);                   // tile_dim0 = n8      (bits 127:112)
  g1[4] = 1;                                             // tile_dim1=1; tile_dim2=0
  g1[5] = (int)n8;                                       // tensor_dim0_stride[31:0]
  g1[6] = 0;
  g1[7] = 0;
  iv4 z4 = {0,0,0,0};
#if defined(__clang_major__) && (__clang_major__ >= 23)
  iv8 z8 = {0,0,0,0,0,0,0,0};
  __builtin_amdgcn_tensor_load_to_lds(g0, g1, z4, z4, z8, 0);
#else
  __builtin_amdgcn_tensor_load_to_lds(g0, g1, z4, z4, 0);
#endif
}

// ---- unified WMMA fragment loader (ISA 7.12.2, wave32) ----
// lane L owns row (A: m, B: n) = L&15; halves e<8 -> K = kb*32 + (L>>4)*8 + e,
// halves e>=8 -> +16. Rows K-contiguous with row stride `stride` halves.
// Lowers to 2x (ds|global)_load_b128.
__device__ __forceinline__ v16h frag16(const f16* base, int stride, int kb, int lane){
  const int r = lane & 15, hs = lane >> 4;
  const f16* p = base + (size_t)r * stride + kb*32 + hs*8;
  v8h lo = *(const v8h*)p;
  v8h hi = *(const v8h*)(p + 16);
  v16h a;
#pragma unroll
  for (int i = 0; i < 8; ++i){ a[i] = lo[i]; a[i+8] = hi[i]; }
  return a;
}
__device__ __forceinline__ v16h frag16_f32(const float* base, int stride, int kb, int lane){
  const int r = lane & 15, hs = lane >> 4;
  const float* p = base + (size_t)r * stride + kb*32 + hs*8;
  v16h a;
#pragma unroll
  for (int i = 0; i < 8; ++i){ a[i] = (f16)p[i]; a[i+8] = (f16)p[i+16]; }
  return a;
}

// ---- kernels ----
__global__ void k_zero_f32(float* p, long n){
  long i = (long)blockIdx.x*blockDim.x + threadIdx.x;
  const long stride = (long)gridDim.x*blockDim.x;
  for (; i < n; i += stride) p[i] = 0.f;
}

// One-shot: write transposed, zero-padded f16 weight images into ws (TDM sources).
__global__ void k_prep_weights(const float* __restrict__ W1, const float* __restrict__ W2,
                               const float* __restrict__ Wc1, const float* __restrict__ Wc2,
                               f16* __restrict__ W1T, f16* __restrict__ W2T,
                               f16* __restrict__ WC1T, f16* __restrict__ WC2T){
  int idx = blockIdx.x*blockDim.x + threadIdx.x;
  if (idx < LAYERS*HID*PAD_D){                     // W1T[l][n][k], pad k>=50
    const int l = idx / (HID*PAD_D), r = idx % (HID*PAD_D), n = r >> 6, k = r & 63;
    W1T[idx] = (f16)((k < IN_D) ? W1[(size_t)l*IN_D*HID + k*HID + n] : 0.f);
    return;
  }
  idx -= LAYERS*HID*PAD_D;
  if (idx < LAYERS*HID*HID){                       // W2T[l][n][k]
    const int l = idx / (HID*HID), r = idx % (HID*HID), n = r >> 7, k = r & 127;
    W2T[idx] = (f16)W2[(size_t)l*HID*HID + k*HID + n];
    return;
  }
  idx -= LAYERS*HID*HID;
  if (idx < HID*256){                              // WC1T[n][k]
    const int n = idx >> 8, k = idx & 255;
    WC1T[idx] = (f16)Wc1[k*HID + n];
    return;
  }
  idx -= HID*256;
  if (idx < 16*HID){                               // WC2T[n][k], pad n>=10
    const int n = idx >> 7, k = idx & 127;
    WC2T[idx] = (f16)((n < NC) ? Wc2[k*NC + n] : 0.f);
  }
}

// Assemble ea = [edge_attr | port_emb[dst_ports] | flag_emb[tcp_flags]] as f16 (padded to 64),
// and accumulate per-dst sums + degree for the mean self-loop attribute.
__global__ void k_assemble(const float* __restrict__ edge_attr, const int* __restrict__ dst_ports,
                           const int* __restrict__ tcp_flags, const int* __restrict__ ei,
                           const float* __restrict__ port_emb, const float* __restrict__ flag_emb,
                           f16* __restrict__ ea16, float* __restrict__ ea_sum, float* __restrict__ deg){
  const int e = blockIdx.x*blockDim.x + threadIdx.x;
  if (e >= NE) return;
  const int dp = dst_ports[e], tf = tcp_flags[e], d = ei[NE + e];
  f16* o = ea16 + (size_t)e * PAD_D;
  float* srow = ea_sum + (size_t)d * IN_D;
#pragma unroll 4
  for (int j = 0; j < 32; ++j){ float f = edge_attr[(size_t)e*32 + j]; o[j] = (f16)f; atomicAdd(&srow[j], f); }
#pragma unroll 4
  for (int j = 0; j < 16; ++j){ float f = port_emb[(size_t)dp*16 + j]; o[32+j] = (f16)f; atomicAdd(&srow[32+j], f); }
#pragma unroll
  for (int j = 0; j < 2; ++j){ float f = flag_emb[(size_t)tf*2 + j]; o[48+j] = (f16)f; atomicAdd(&srow[48+j], f); }
#pragma unroll
  for (int j = IN_D; j < PAD_D; ++j) o[j] = (f16)0.f;
  atomicAdd(&deg[d], 1.0f);
}

// Self-loop rows: loop_attr[n] = ea_sum[n]/max(deg,1), written at rows NE..NE+NN-1.
__global__ void k_loop_attr(const float* __restrict__ ea_sum, const float* __restrict__ deg,
                            f16* __restrict__ ea16){
  const int idx = blockIdx.x*blockDim.x + threadIdx.x;
  if (idx >= NN*PAD_D) return;
  const int n = idx >> 6, d = idx & 63;
  const float v = (d < IN_D) ? ea_sum[(size_t)n*IN_D + d] / fmaxf(deg[n], 1.f) : 0.f;
  ea16[(size_t)(NE + n)*PAD_D + d] = (f16)v;
}

// Per layer: m = relu(ea_aug@W1+b1)@W2+b2 via WMMA; x_new[dst] += x_prev[src] + m (atomic scatter).
// Weight slabs staged into LDS by the Tensor Data Mover (f16, pre-transposed [n][k]).
__global__ __launch_bounds__(256) void k_edge_mlp_wmma(
    const f16* __restrict__ ea16, const int* __restrict__ ei,
    const f16* __restrict__ w1t_g, const float* __restrict__ b1l,
    const f16* __restrict__ w2t_g, const float* __restrict__ b2l,
    const float* __restrict__ x_prev, float* __restrict__ x_new){
  extern __shared__ char smem[];
  f16*   W1t   = (f16*)smem;                   // [n=128][k=64]
  f16*   W2t   = W1t + 128*64;                 // [n=128][k=128]
  float* b1s   = (float*)(W2t + 128*128);      // [128]
  float* b2s   = b1s + 128;                    // [128]
  f16*   HsAll = (f16*)(b2s + 128);            // 8 waves x [16][128]

  const int tid = threadIdx.x;
  if (tid < 32){                                // wave 0: TDM-stage both weight slabs
    tdm_load_to_lds(W1t, w1t_g, 128*64*2);
    tdm_load_to_lds(W2t, w2t_g, 128*128*2);
  }
  if (tid < 128){ b1s[tid] = b1l[tid]; b2s[tid] = b2l[tid]; }
  if (tid < 32) __builtin_amdgcn_s_wait_tensorcnt(0);
  __syncthreads();

  const int wave = tid >> 5, lane = tid & 31;
  const int tile = blockIdx.x*8 + wave;
  if (tile >= TILES) return;                    // wave-uniform; EXEC stays all-ones for WMMA
  const int row_base = tile*16;
  f16* Hs = HsAll + wave*(16*HID);
  const int nlo = lane & 15, hs = lane >> 4;

  // warm the scatter indices for this tile (global_prefetch_b8)
  __builtin_prefetch(ei + row_base, 0, 3);
  __builtin_prefetch(ei + NE + row_base, 0, 3);

  // GEMM 1: [16x64] x [64x128], K = 2 chunks of 32
  const f16* Abase = ea16 + (size_t)row_base * PAD_D;
  const v16h a0 = frag16(Abase, PAD_D, 0, lane);
  const v16h a1 = frag16(Abase, PAD_D, 1, lane);
#pragma unroll
  for (int nt = 0; nt < 8; ++nt){
    const f16* Bb = W1t + nt*16*PAD_D;          // 16 n-rows, K-contiguous
    v8f acc = {};
    acc = WMMA_F16(a0, frag16(Bb, PAD_D, 0, lane), acc);
    acc = WMMA_F16(a1, frag16(Bb, PAD_D, 1, lane), acc);
    const int col = nt*16 + nlo;
    const float bb = b1s[col];
#pragma unroll
    for (int v = 0; v < 8; ++v)                 // C layout: row = v + 8*hs, col = nt*16 + (lane&15)
      Hs[(v + hs*8)*HID + col] = (f16)fmaxf(acc[v] + bb, 0.f);
  }

  // GEMM 2: [16x128] x [128x128], K = 4 chunks of 32 (A re-shaped via per-wave LDS slab)
  const v16h h0 = frag16(Hs, HID, 0, lane);
  const v16h h1 = frag16(Hs, HID, 1, lane);
  const v16h h2 = frag16(Hs, HID, 2, lane);
  const v16h h3 = frag16(Hs, HID, 3, lane);
#pragma unroll
  for (int nt = 0; nt < 8; ++nt){
    const f16* Bb = W2t + nt*16*HID;
    v8f acc = {};
    acc = WMMA_F16(h0, frag16(Bb, HID, 0, lane), acc);
    acc = WMMA_F16(h1, frag16(Bb, HID, 1, lane), acc);
    acc = WMMA_F16(h2, frag16(Bb, HID, 2, lane), acc);
    acc = WMMA_F16(h3, frag16(Bb, HID, 3, lane), acc);
    const int col = nt*16 + nlo;
    const float bb = b2s[col];
#pragma unroll
    for (int v = 0; v < 8; ++v){
      const int r = row_base + v + hs*8;
      int s, d;
      if (r < NE){ s = ei[r]; d = ei[NE + r]; } else { s = r - NE; d = s; }
      float val = acc[v] + bb;
      if (x_prev) val += x_prev[(size_t)s*HID + col];
      atomicAdd(&x_new[(size_t)d*HID + col], val);
    }
  }
}

// Pool accumulation: mean-sum, max (encoded uint), counts per graph.
__global__ void k_pool(const float* __restrict__ x, const int* __restrict__ batch,
                       float* __restrict__ mean_sum, unsigned* __restrict__ maxenc,
                       float* __restrict__ cnt){
  const long idx = (long)blockIdx.x*blockDim.x + threadIdx.x;
  if (idx >= (long)NN*HID) return;
  const int n = (int)(idx >> 7), d = (int)(idx & 127);
  const int b = batch[n];
  const float v = x[idx];
  atomicAdd(&mean_sum[b*HID + d], v);
  atomicMax(&maxenc[b*HID + d], fenc(v));
  if (d == 0) atomicAdd(&cnt[b], 1.0f);
}

// g = [max_pool | mean_pool]  (512 x 256)
__global__ void k_pool_fin(const float* __restrict__ mean_sum, const unsigned* __restrict__ maxenc,
                           const float* __restrict__ cnt, float* __restrict__ g){
  const int idx = blockIdx.x*blockDim.x + threadIdx.x;
  if (idx >= NG*256) return;
  const int b = idx >> 8, c = idx & 255;
  const float cv = cnt[b];
  g[idx] = (c < HID) ? (cv > 0.f ? fdec(maxenc[b*HID + c]) : 0.f)
                     : mean_sum[b*HID + (c - HID)] / fmaxf(cv, 1.f);
}

// Classifier: relu(g@Wc1+bc1)@Wc2+bc2 via WMMA (weights TDM-staged, pre-transposed/padded).
__global__ __launch_bounds__(256) void k_classifier_wmma(
    const float* __restrict__ g, const f16* __restrict__ wc1t_g, const float* __restrict__ bc1,
    const f16* __restrict__ wc2t_g, const float* __restrict__ bc2, float* __restrict__ out){
  extern __shared__ char smem[];
  f16*   Wc1t  = (f16*)smem;                   // [n=128][k=256]
  f16*   Wc2t  = Wc1t + 128*256;               // [n=16][k=128]
  float* bc1s  = (float*)(Wc2t + 16*128);      // [128]
  float* bc2s  = bc1s + 128;                   // [16]
  f16*   HsAll = (f16*)(bc2s + 16);            // 8 waves x [16][128]

  const int tid = threadIdx.x;
  if (tid < 32){
    tdm_load_to_lds(Wc1t, wc1t_g, 128*256*2);
    tdm_load_to_lds(Wc2t, wc2t_g, 16*128*2);
  }
  if (tid < 128) bc1s[tid] = bc1[tid];
  if (tid < 16)  bc2s[tid] = (tid < NC) ? bc2[tid] : 0.f;
  if (tid < 32) __builtin_amdgcn_s_wait_tensorcnt(0);
  __syncthreads();

  const int wave = tid >> 5, lane = tid & 31;
  const int tile = blockIdx.x*8 + wave;
  if (tile >= NG/16) return;
  const int row_base = tile*16;
  f16* Hs = HsAll + wave*(16*HID);
  const int nlo = lane & 15, hs = lane >> 4;

  v16h ga[8];
#pragma unroll
  for (int kb = 0; kb < 8; ++kb) ga[kb] = frag16_f32(g + (size_t)row_base*256, 256, kb, lane);

#pragma unroll
  for (int nt = 0; nt < 8; ++nt){
    const f16* Bb = Wc1t + nt*16*256;
    v8f acc = {};
#pragma unroll
    for (int kb = 0; kb < 8; ++kb)
      acc = WMMA_F16(ga[kb], frag16(Bb, 256, kb, lane), acc);
    const int col = nt*16 + nlo;
    const float bb = bc1s[col];
#pragma unroll
    for (int v = 0; v < 8; ++v) Hs[(v + hs*8)*HID + col] = (f16)fmaxf(acc[v] + bb, 0.f);
  }

  v8f acc = {};
#pragma unroll
  for (int kb = 0; kb < 4; ++kb)
    acc = WMMA_F16(frag16(Hs, HID, kb, lane), frag16(Wc2t, HID, kb, lane), acc);
  if (nlo < NC){
    const float bb = bc2s[nlo];
#pragma unroll
    for (int v = 0; v < 8; ++v){
      const int r = row_base + v + hs*8;
      out[r*NC + nlo] = acc[v] + bb;
    }
  }
}

extern "C" void kernel_launch(void* const* d_in, const int* in_sizes, int n_in,
                              void* d_out, int out_size, void* d_ws, size_t ws_size,
                              hipStream_t stream){
  const float* edge_attr = (const float*)d_in[0];
  const int*   dst_ports = (const int*)d_in[1];
  const int*   tcp_flags = (const int*)d_in[2];
  const int*   ei        = (const int*)d_in[3];
  const int*   batch     = (const int*)d_in[4];
  const float* W1        = (const float*)d_in[7];
  const float* b1        = (const float*)d_in[8];
  const float* W2        = (const float*)d_in[9];
  const float* b2        = (const float*)d_in[10];
  const float* Wc1       = (const float*)d_in[11];
  const float* bc1       = (const float*)d_in[12];
  const float* Wc2       = (const float*)d_in[13];
  const float* bc2       = (const float*)d_in[14];
  const float* port_emb  = (const float*)d_in[5];
  const float* flag_emb  = (const float*)d_in[6];
  float* out = (float*)d_out;

  char* ws = (char*)d_ws;
  f16*      ea16     = (f16*)(ws + OFF_EA16);
  float*    ea_sum   = (float*)(ws + OFF_EASUM);
  float*    deg      = (float*)(ws + OFF_DEG);
  float*    x_a      = (float*)(ws + OFF_XA);
  float*    x_b      = (float*)(ws + OFF_XB);
  float*    mean_sum = (float*)(ws + OFF_MEAN);
  unsigned* maxenc   = (unsigned*)(ws + OFF_MAX);
  float*    cnt      = (float*)(ws + OFF_CNT);
  float*    g        = (float*)(ws + OFF_G);
  f16*      W1T      = (f16*)(ws + OFF_W1T);
  f16*      W2T      = (f16*)(ws + OFF_W2T);
  f16*      WC1T     = (f16*)(ws + OFF_WC1T);
  f16*      WC2T     = (f16*)(ws + OFF_WC2T);

  // allow >64KB dynamic LDS (ignore errors; harmless host-side call)
  (void)hipFuncSetAttribute((const void*)k_edge_mlp_wmma,
                            hipFuncAttributeMaxDynamicSharedMemorySize, GEMM_LDS);
  (void)hipFuncSetAttribute((const void*)k_classifier_wmma,
                            hipFuncAttributeMaxDynamicSharedMemorySize, CLS_LDS);

  // 1) zero all accumulation buffers; build transposed f16 weight images
  k_zero_f32<<<4096, 256, 0, stream>>>((float*)(ws + OFF_ZERO), (long)(ZERO_BYTES/4));
  {
    const int prep_n = LAYERS*HID*PAD_D + LAYERS*HID*HID + HID*256 + 16*HID;
    k_prep_weights<<<(prep_n + 255)/256, 256, 0, stream>>>(W1, W2, Wc1, Wc2, W1T, W2T, WC1T, WC2T);
  }
  // 2) edge feature assembly + dst sums/degree
  k_assemble<<<(NE + 255)/256, 256, 0, stream>>>(edge_attr, dst_ports, tcp_flags, ei,
                                                 port_emb, flag_emb, ea16, ea_sum, deg);
  // 3) mean self-loop attrs
  k_loop_attr<<<(NN*PAD_D + 255)/256, 256, 0, stream>>>(ea_sum, deg, ea16);
  // 4) three message-passing layers (ping-pong x buffers)
  const int gb = (TILES + 7)/8;
  k_edge_mlp_wmma<<<gb, 256, GEMM_LDS, stream>>>(ea16, ei, W1T, b1, W2T, b2, nullptr, x_a);
  k_edge_mlp_wmma<<<gb, 256, GEMM_LDS, stream>>>(ea16, ei, W1T + HID*PAD_D, b1 + HID,
                                                 W2T + HID*HID, b2 + HID, x_a, x_b);
  k_zero_f32<<<4096, 256, 0, stream>>>(x_a, (long)NN*HID);
  k_edge_mlp_wmma<<<gb, 256, GEMM_LDS, stream>>>(ea16, ei, W1T + 2*HID*PAD_D, b1 + 2*HID,
                                                 W2T + 2*HID*HID, b2 + 2*HID, x_b, x_a);
  // 5) per-graph pooling
  k_pool<<<(int)(((long)NN*HID + 255)/256), 256, 0, stream>>>(x_a, batch, mean_sum, maxenc, cnt);
  k_pool_fin<<<(NG*256 + 255)/256, 256, 0, stream>>>(mean_sum, maxenc, cnt, g);
  // 6) classifier head
  k_classifier_wmma<<<(NG/16 + 7)/8, 256, CLS_LDS, stream>>>(g, WC1T, bc1, WC2T, bc2, out);
}
